// CrossModalityAttention_21337397527144
// MI455X (gfx1250) — matrix operand order repaired and checked by reading, not compiled, not run on hardware
//
#include <hip/hip_runtime.h>
#include <stdint.h>

#define HD   1024
#define KMOD 8
#define HDIM 128

typedef __attribute__((ext_vector_type(16))) __bf16        v16bf;
typedef __attribute__((ext_vector_type(8)))  float         v8f;
typedef __attribute__((ext_vector_type(4)))  unsigned int  v4u;

union FragBF {
    v16bf          v;
    v4u            q[2];
    unsigned short s[16];
};

__device__ __forceinline__ unsigned short f2bf(float f) {
    unsigned int u = __float_as_uint(f);
    unsigned int r = (u + 0x7fffu + ((u >> 16) & 1u)) >> 16;   // round-to-nearest-even
    return (unsigned short)r;
}
__device__ __forceinline__ float bf2f(unsigned short h) {
    return __uint_as_float(((unsigned int)h) << 16);
}

// ---------------------------------------------------------------------------
// Kernel 1: convert the four [1024x1024] fp32 weight matrices to bf16 in ws.
// Layout in ws: [wq | wk | wv | wo], each 1M ushort, row-major [out, in].
// ---------------------------------------------------------------------------
__global__ void convert_weights(const float* __restrict__ wq,
                                const float* __restrict__ wk,
                                const float* __restrict__ wv,
                                const float* __restrict__ wo,
                                unsigned short* __restrict__ out) {
    int idx = blockIdx.x * blockDim.x + threadIdx.x;       // 0 .. 4M-1
    const int N = HD * HD;                                 // 2^20
    if (idx >= 4 * N) return;
    int w = idx >> 20;
    int i = idx & (N - 1);
    const float* src = (w == 0) ? wq : (w == 1) ? wk : (w == 2) ? wv : wo;
    out[idx] = f2bf(src[i]);
}

// ---------------------------------------------------------------------------
// Kernel 2: fully fused cross-modality attention block.
// One block = 32 rows (4 batches) of the [B*K, D] view; 512 threads = 16 waves.
// wave w: M-group mg = w>>3 (rows 16*mg..+16), N-slice nw = w&7 (cols 128*nw..+128).
// LDS (256 KB dynamic):
//   [0      ,  65536) sq   bf16 32x1024   | sres f32 32x1024 (phase>=3, + sk region)
//   [65536  , 131072) sk   bf16 32x1024   |
//   [131072 , 196608) sv   bf16 32x1024   | sred/sstat (phase 4)
//   [196608 , 262144) sx   bf16 32x1024 (phase 0/1) -> sao (phase 2/3)
// ---------------------------------------------------------------------------
__global__ __launch_bounds__(512) void fused_attn(
        const float* __restrict__ x,        // [B,K,D]
        const float* __restrict__ mask,     // [B,K]
        const unsigned short* __restrict__ wbf,  // [4][1024][1024] bf16
        const float* __restrict__ bq, const float* __restrict__ bk,
        const float* __restrict__ bv, const float* __restrict__ bo,
        const float* __restrict__ rel,      // [K,K]
        const float* __restrict__ gamma, const float* __restrict__ beta,
        float* __restrict__ out)            // [B,K,D]
{
    extern __shared__ unsigned char smem[];
    unsigned short* sq   = (unsigned short*)(smem);
    unsigned short* sk   = (unsigned short*)(smem + 65536);
    unsigned short* sv   = (unsigned short*)(smem + 131072);
    unsigned short* sx   = (unsigned short*)(smem + 196608);  // x bf16, later sao
    unsigned short* sao  = (unsigned short*)(smem + 196608);
    float*          sres = (float*)(smem);                    // 32x1024 f32, phase>=3
    float*          sred = (float*)(smem + 131072);           // 32x16x2 f32, phase 4
    float*          sstat= (float*)(smem + 135168);           // 32x2 f32

    const int t     = threadIdx.x;
    const int lane  = t & 31;
    const int wave  = t >> 5;                 // 0..15
    const int mg    = wave >> 3;              // 0..1  M-group
    const int nwave = (wave & 7) * 128;       // N-slice base
    const int mbase = mg * 16;
    const int row0  = blockIdx.x * 32;        // row in [B*K, D] view
    const int mloc  = lane & 15;
    const int half  = lane >> 4;              // K-split per ISA 16-bit layouts
    const int mrow0 = half * 8;               // C/D: VGPR r -> M = r + 8*half

    // ------------- Phase 0: stage x (fp32 -> bf16) into LDS once
    {
        int row = t >> 4, seg = t & 15;       // 32 rows x 16 segments of 64
        const float*  xp = x + (size_t)(row0 + row) * HD + seg * 64;
        unsigned int* sp = (unsigned int*)(sx + row * HD + seg * 64);
        #pragma unroll 8
        for (int i = 0; i < 32; ++i) {
            float lo = xp[2 * i], hi = xp[2 * i + 1];
            sp[i] = (unsigned int)f2bf(lo) | ((unsigned int)f2bf(hi) << 16);
        }
    }
    __syncthreads();

    // ------------- Phase 1: Q/K/V projections  y = x @ W^T + b  (bf16 WMMA)
    const float*    biases[3] = { bq, bk, bv };
    unsigned short* dsts[3]   = { sq, sk, sv };

    for (int p = 0; p < 3; ++p) {
        const unsigned short* W = wbf + (size_t)p * (HD * HD);
        const unsigned short* Wrow = W + (size_t)(nwave + mloc) * HD + half * 16;
        v8f acc[8];
        #pragma unroll
        for (int nt = 0; nt < 8; ++nt)
            #pragma unroll
            for (int r = 0; r < 8; ++r) acc[nt][r] = 0.f;

        for (int kb = 0; kb < HD; kb += 32) {
            // all 8 B fragments first: 16 loads in flight, one wait point
            FragBF b[8];
            #pragma unroll
            for (int nt = 0; nt < 8; ++nt) {
                const v4u* pb = (const v4u*)(Wrow + (size_t)nt * 16 * HD + kb);
                b[nt].q[0] = pb[0];
                b[nt].q[1] = pb[1];
            }
            FragBF a;   // A fragment from staged x (two ds_load_b128)
            a.q[0] = *(const v4u*)(sx + (mbase + mloc) * HD + kb + half * 16);
            a.q[1] = *(const v4u*)(sx + (mbase + mloc) * HD + kb + half * 16 + 8);
            #pragma unroll
            for (int nt = 0; nt < 8; ++nt)
                acc[nt] = __builtin_amdgcn_wmma_f32_16x16x32_bf16(
                              false, a.v, false, b[nt].v, (short)0, acc[nt], false, false);
            if (kb + 64 < HD)   // stream weights ahead: gfx1250 global_prefetch_b8
                __builtin_prefetch(Wrow + kb + 64, 0, 1);
        }
        // bias add + bf16 store to LDS
        const float*    bias = biases[p];
        unsigned short* dst  = dsts[p];
        #pragma unroll
        for (int nt = 0; nt < 8; ++nt) {
            int   n  = nwave + nt * 16 + mloc;
            float bb = bias[n];
            #pragma unroll
            for (int r = 0; r < 8; ++r)
                dst[(mbase + mrow0 + r) * HD + n] = f2bf(acc[nt][r] + bb);
        }
    }
    __syncthreads();

    // ------------- Phase 2: 8x8 attention per (batch, head, query); 256 threads
    if (t < 256) {
        int bl = t >> 6;            // local batch 0..3
        int h  = (t >> 3) & 7;      // head
        int qi = t & 7;             // query modality
        int bg = blockIdx.x * 4 + bl;
        int rq = bl * 8 + qi;
        const unsigned short* qv = sq + rq * HD + h * HDIM;

        float sc[8];
        #pragma unroll
        for (int j = 0; j < 8; ++j) sc[j] = 0.f;
        for (int d = 0; d < HDIM; ++d) {
            float qd = bf2f(qv[d]);
            #pragma unroll
            for (int j = 0; j < 8; ++j)
                sc[j] += qd * bf2f(sk[(bl * 8 + j) * HD + h * HDIM + d]);
        }
        const float isq = 0.088388347648318447f;   // 1/sqrt(128)
        float mq = mask[bg * KMOD + qi];
        float mx = -__builtin_inff();
        #pragma unroll
        for (int j = 0; j < 8; ++j) {
            float s  = sc[j] * isq + rel[qi * KMOD + j];
            float mj = mask[bg * KMOD + j];
            s = (mq * mj > 0.f) ? s : -__builtin_inff();
            sc[j] = s;
            mx = fmaxf(mx, s);
        }
        float ssum = 0.f;
        #pragma unroll
        for (int j = 0; j < 8; ++j) { float e = __expf(sc[j] - mx); sc[j] = e; ssum += e; }
        float rs = 1.f / ssum;
        #pragma unroll
        for (int j = 0; j < 8; ++j) sc[j] *= rs;

        unsigned short* ao = sao + rq * HD + h * HDIM;   // overwrites sx (dead)
        for (int d = 0; d < HDIM; ++d) {
            float o = 0.f;
            #pragma unroll
            for (int j = 0; j < 8; ++j)
                o += sc[j] * bf2f(sv[(bl * 8 + j) * HD + h * HDIM + d]);
            ao[d] = f2bf(o);
        }
    }
    __syncthreads();

    // ------------- Phase 3: output projection + residual (A from LDS bf16)
    {
        const unsigned short* W = wbf + (size_t)3 * (HD * HD);
        const unsigned short* Wrow = W + (size_t)(nwave + mloc) * HD + half * 16;
        v8f acc[8];
        #pragma unroll
        for (int nt = 0; nt < 8; ++nt)
            #pragma unroll
            for (int r = 0; r < 8; ++r) acc[nt][r] = 0.f;

        for (int kb = 0; kb < HD; kb += 32) {
            FragBF b[8];
            #pragma unroll
            for (int nt = 0; nt < 8; ++nt) {
                const v4u* pb = (const v4u*)(Wrow + (size_t)nt * 16 * HD + kb);
                b[nt].q[0] = pb[0];
                b[nt].q[1] = pb[1];
            }
            FragBF a;
            a.q[0] = *(const v4u*)(sao + (mbase + mloc) * HD + kb + half * 16);
            a.q[1] = *(const v4u*)(sao + (mbase + mloc) * HD + kb + half * 16 + 8);
            #pragma unroll
            for (int nt = 0; nt < 8; ++nt)
                acc[nt] = __builtin_amdgcn_wmma_f32_16x16x32_bf16(
                              false, a.v, false, b[nt].v, (short)0, acc[nt], false, false);
            if (kb + 64 < HD)
                __builtin_prefetch(Wrow + kb + 64, 0, 1);
        }
        #pragma unroll
        for (int nt = 0; nt < 8; ++nt) {
            int   n  = nwave + nt * 16 + mloc;
            float bb = bo[n];
            #pragma unroll
            for (int r = 0; r < 8; ++r) {
                int m = mbase + mrow0 + r;
                sres[m * HD + n] = acc[nt][r] + bb + x[(size_t)(row0 + m) * HD + n];
            }
        }
    }
    __syncthreads();

    // ------------- Phase 4: LayerNorm (32 rows x 16 segs of 64)
    {
        int row = t >> 4, seg = t & 15;
        const float* rp = sres + row * HD + seg * 64;
        float s1 = 0.f, s2 = 0.f;
        #pragma unroll 8
        for (int i = 0; i < 64; ++i) { float v = rp[i]; s1 += v; s2 += v * v; }
        sred[(row * 16 + seg) * 2 + 0] = s1;
        sred[(row * 16 + seg) * 2 + 1] = s2;
        __syncthreads();
        if (seg == 0) {
            float S1 = 0.f, S2 = 0.f;
            #pragma unroll
            for (int i = 0; i < 16; ++i) {
                S1 += sred[(row * 16 + i) * 2 + 0];
                S2 += sred[(row * 16 + i) * 2 + 1];
            }
            float mu  = S1 * (1.f / 1024.f);
            float var = S2 * (1.f / 1024.f) - mu * mu;
            sstat[row * 2 + 0] = mu;
            sstat[row * 2 + 1] = rsqrtf(var + 1e-5f);
        }
        __syncthreads();
        float mu  = sstat[row * 2 + 0];
        float rsd = sstat[row * 2 + 1];
        float* op = out + (size_t)(row0 + row) * HD + seg * 64;
        const float* gp = gamma + seg * 64;
        const float* bp = beta  + seg * 64;
        #pragma unroll 8
        for (int i = 0; i < 64; ++i)
            op[i] = (rp[i] - mu) * rsd * gp[i] + bp[i];
    }
}

extern "C" void kernel_launch(void* const* d_in, const int* in_sizes, int n_in,
                              void* d_out, int out_size, void* d_ws, size_t ws_size,
                              hipStream_t stream) {
    const float* x     = (const float*)d_in[0];
    const float* mask  = (const float*)d_in[1];
    const float* wq    = (const float*)d_in[2];
    const float* bq    = (const float*)d_in[3];
    const float* wk    = (const float*)d_in[4];
    const float* bk    = (const float*)d_in[5];
    const float* wv    = (const float*)d_in[6];
    const float* bv    = (const float*)d_in[7];
    const float* wo    = (const float*)d_in[8];
    const float* bo    = (const float*)d_in[9];
    const float* rel   = (const float*)d_in[10];
    const float* gamma = (const float*)d_in[11];
    const float* beta  = (const float*)d_in[12];
    float* out = (float*)d_out;
    unsigned short* wbf = (unsigned short*)d_ws;   // 8 MB bf16 weights

    int n = 4 * HD * HD;
    convert_weights<<<(n + 255) / 256, 256, 0, stream>>>(wq, wk, wv, wo, wbf);

    const int B = 8192;
    fused_attn<<<dim3(B * KMOD / 32), dim3(512), 262144, stream>>>(
        x, mask, wbf, bq, bk, bv, bo, rel, gamma, beta, out);
}
